// GraphEncoder_76020921139986
// MI455X (gfx1250) — matrix operand order repaired and checked by reading, not compiled, not run on hardware
//
#include <hip/hip_runtime.h>
#include <stdint.h>

#define IN_F  256
#define HIDF  128
#define OUTF  256
#define FINF  384

typedef __attribute__((ext_vector_type(8)))  __bf16 v8bf;
typedef __attribute__((ext_vector_type(16))) __bf16 v16bf;
typedef __attribute__((ext_vector_type(8)))  float  v8f;

// ---------- helpers ----------
__device__ __forceinline__ unsigned short f32_to_bf16_bits(float f) {
    union { float f; unsigned int u; } v; v.f = f;
    unsigned int u = v.u;
    u += 0x7FFFu + ((u >> 16) & 1u);      // round to nearest even
    return (unsigned short)(u >> 16);
}
__device__ __forceinline__ float bf16_bits_to_f32(unsigned short h) {
    union { unsigned int u; float f; } v; v.u = ((unsigned int)h) << 16;
    return v.f;
}
__device__ __forceinline__ void atomAddF32(float* p, float v) {
    __hip_atomic_fetch_add(p, v, __ATOMIC_RELAXED, __HIP_MEMORY_SCOPE_AGENT);
}

// A fragment (16x32 bf16, MxK): lane<16 -> M=lane, K = {k0+half*8..+7} U {k0+16+half*8..+7}
__device__ __forceinline__ v16bf load_a_frag(const unsigned short* base, int row, int ld,
                                             int k0, int half) {
    const __bf16* p = reinterpret_cast<const __bf16*>(base + (size_t)row * ld + k0 + half * 8);
    v8bf lo = *reinterpret_cast<const v8bf*>(p);
    v8bf hi = *reinterpret_cast<const v8bf*>(p + 16);
    return __builtin_shufflevector(lo, hi, 0,1,2,3,4,5,6,7,8,9,10,11,12,13,14,15);
}
// B fragment (32x16 bf16, KxN), B[k][n] = W[n][k]: lane holds col n=lane&15,
// K = {k0+half*8..+7} U {k0+16+half*8..+7} -> two 16B chunks of W row.
__device__ __forceinline__ v16bf load_b_frag(const unsigned short* W, int col, int ld,
                                             int k0, int half) {
    const __bf16* p = reinterpret_cast<const __bf16*>(W + (size_t)col * ld + k0 + half * 8);
    v8bf lo = *reinterpret_cast<const v8bf*>(p);
    v8bf hi = *reinterpret_cast<const v8bf*>(p + 16);
    return __builtin_shufflevector(lo, hi, 0,1,2,3,4,5,6,7,8,9,10,11,12,13,14,15);
}

// ---------- elementwise kernels ----------
__global__ void zero_f32_kernel(float* p, size_t n) {
    size_t i = (size_t)blockIdx.x * blockDim.x + threadIdx.x;
    if (i < n) p[i] = 0.0f;
}
__global__ void cvt_f32_to_bf16_kernel(const float* in, unsigned short* out, size_t n) {
    size_t i = (size_t)blockIdx.x * blockDim.x + threadIdx.x;
    if (i < n) out[i] = f32_to_bf16_bits(in[i]);
}
__global__ void finalize_mean_kernel(const float* agg, const float* deg,
                                     unsigned short* outB, size_t total, int F) {
    size_t i = (size_t)blockIdx.x * blockDim.x + threadIdx.x;
    if (i >= total) return;
    int row = (int)(i / F);
    float inv = 1.0f / fmaxf(deg[row], 1.0f);
    outB[i] = f32_to_bf16_bits(agg[i] * inv);
}

// ---------- scatter (segment-sum) kernels: one wave per edge ----------
__global__ void scatter_f32_kernel(const float* X, const long long* src, const long long* dst,
                                   float* agg, float* deg, int E, int F) {
    int warp = (int)(((size_t)blockIdx.x * blockDim.x + threadIdx.x) >> 5);
    int lane = threadIdx.x & 31;
    if (warp >= E) return;
    long long s = src[warp], d = dst[warp];
    const float* xs = X + (size_t)s * F;
    float* ad = agg + (size_t)d * F;
    for (int f = lane; f < F; f += 32) atomAddF32(&ad[f], xs[f]);
    if (lane == 0) atomAddF32(&deg[d], 1.0f);
}
__global__ void scatter_bf16_kernel(const unsigned short* Xb, const long long* src,
                                    const long long* dst, float* agg, int E, int F) {
    int warp = (int)(((size_t)blockIdx.x * blockDim.x + threadIdx.x) >> 5);
    int lane = threadIdx.x & 31;
    if (warp >= E) return;
    long long s = src[warp], d = dst[warp];
    const unsigned short* xs = Xb + (size_t)s * F;
    float* ad = agg + (size_t)d * F;
    for (int f = lane; f < F; f += 32) atomAddF32(&ad[f], bf16_bits_to_f32(xs[f]));
}

// ---------- WMMA GEMM ----------
// out[M,N] = act( A1[M,K1] @ W1[N,K1]^T + A2[M,K2] @ W2[N,K2]^T + bias )
// One wave per 16(M) x 64(N) strip; A frag reused across 4 B tiles per K-step.
__global__ __launch_bounds__(256) void gemm_bf16_wmma_kernel(
    const unsigned short* __restrict__ A1, const unsigned short* __restrict__ W1, int K1,
    const unsigned short* __restrict__ A2, const unsigned short* __restrict__ W2, int K2,
    const float* __restrict__ bias, int M, int N,
    float* __restrict__ outF, unsigned short* __restrict__ outB, int relu) {
    const int wavesPerBlock = blockDim.x >> 5;
    int wave   = blockIdx.x * wavesPerBlock + (threadIdx.x >> 5);
    int nStrips = N >> 6;                 // 64-wide strips
    int mTiles  = M >> 4;
    if (wave >= mTiles * nStrips) return; // wave-uniform: EXEC stays all-1s
    int mt = wave / nStrips;
    int ns = wave % nStrips;
    int lane = threadIdx.x & 31;
    int half = lane >> 4;
    int r    = lane & 15;
    int mBase = mt << 4;
    int nBase = ns << 6;

    v8f acc[4];
#pragma unroll
    for (int t = 0; t < 4; ++t) {
        float b = bias ? bias[nBase + t * 16 + r] : 0.0f;
#pragma unroll
        for (int i = 0; i < 8; ++i) acc[t][i] = b;
    }

    const unsigned short* As = A1;
    const unsigned short* Ws = W1;
    int K = K1;
#pragma unroll 1
    for (int srcIdx = 0; srcIdx < 2; ++srcIdx) {
        if (As) {
            for (int k0 = 0; k0 < K; k0 += 32) {
                v16bf a = load_a_frag(As, mBase + r, K, k0, half);
#pragma unroll
                for (int t = 0; t < 4; ++t) {
                    v16bf b = load_b_frag(Ws, nBase + t * 16 + r, K, k0, half);
                    acc[t] = __builtin_amdgcn_wmma_f32_16x16x32_bf16(
                        /*neg_a=*/false, a, /*neg_b=*/false, b,
                        /*c_mod=*/(short)0, acc[t],
                        /*reuse_a=*/false, /*reuse_b=*/false);
                }
            }
        }
        As = A2; Ws = W2; K = K2;
    }

    // C/D layout: elem i -> row = mBase + half*8 + i, col = nBase + t*16 + r
#pragma unroll
    for (int t = 0; t < 4; ++t) {
        int col = nBase + t * 16 + r;
#pragma unroll
        for (int i = 0; i < 8; ++i) {
            int row = mBase + half * 8 + i;
            float v = acc[t][i];
            if (relu) v = fmaxf(v, 0.0f);
            if (outF) outF[(size_t)row * N + col] = v;
            else      outB[(size_t)row * N + col] = f32_to_bf16_bits(v);
        }
    }
}

// ---------- host orchestration ----------
static inline int ceil_div_i(long long a, long long b) { return (int)((a + b - 1) / b); }

extern "C" void kernel_launch(void* const* d_in, const int* in_sizes, int n_in,
                              void* d_out, int out_size, void* d_ws, size_t ws_size,
                              hipStream_t stream) {
    const float*     x    = (const float*)d_in[0];
    const long long* ei   = (const long long*)d_in[1];   // int64 [2, E]
    const float*     W1l  = (const float*)d_in[2];
    const float*     b1   = (const float*)d_in[3];
    const float*     W1r  = (const float*)d_in[4];
    const float*     W2l  = (const float*)d_in[5];
    const float*     b2   = (const float*)d_in[6];
    const float*     W2r  = (const float*)d_in[7];
    const float*     Wp   = (const float*)d_in[8];
    const float*     bp   = (const float*)d_in[9];

    const int Nn = in_sizes[0] / IN_F;     // 50000 (multiple of 16)
    const int E  = in_sizes[1] / 2;        // 800000
    const long long* src = ei;
    const long long* dst = ei + E;

    // workspace carve-out (256B aligned)
    size_t off = 0;
    char* ws = (char*)d_ws;
    auto take = [&](size_t bytes) -> void* {
        void* p = (void*)(ws + off);
        off += (bytes + 255) & ~(size_t)255;
        return p;
    };
    float*          aggF = (float*)take((size_t)Nn * IN_F * 4);  // reused as [Nn,128] f32
    float*          deg  = (float*)take((size_t)Nn * 4);
    unsigned short* xB   = (unsigned short*)take((size_t)Nn * IN_F * 2);
    unsigned short* aggB = (unsigned short*)take((size_t)Nn * IN_F * 2); // reused [Nn,128]
    unsigned short* h1B  = (unsigned short*)take((size_t)Nn * HIDF * 2);
    unsigned short* h2B  = (unsigned short*)take((size_t)Nn * OUTF * 2);
    unsigned short* w1lB = (unsigned short*)take((size_t)HIDF * IN_F * 2);
    unsigned short* w1rB = (unsigned short*)take((size_t)HIDF * IN_F * 2);
    unsigned short* w2lB = (unsigned short*)take((size_t)OUTF * HIDF * 2);
    unsigned short* w2rB = (unsigned short*)take((size_t)OUTF * HIDF * 2);
    unsigned short* wpB  = (unsigned short*)take((size_t)FINF * OUTF * 2);
    (void)ws_size; (void)n_in;

    const int T = 256;
    auto cvt = [&](const float* in, unsigned short* out, size_t n) {
        cvt_f32_to_bf16_kernel<<<ceil_div_i((long long)n, T), T, 0, stream>>>(in, out, n);
    };

    // 0) precision staging: weights + x -> bf16
    cvt(W1l, w1lB, (size_t)HIDF * IN_F);
    cvt(W1r, w1rB, (size_t)HIDF * IN_F);
    cvt(W2l, w2lB, (size_t)OUTF * HIDF);
    cvt(W2r, w2rB, (size_t)OUTF * HIDF);
    cvt(Wp,  wpB,  (size_t)FINF * OUTF);
    cvt(x,   xB,   (size_t)Nn * IN_F);

    // 1) conv1 aggregation: agg1 = segment_sum(x[src], dst), deg = segment_sum(1, dst)
    zero_f32_kernel<<<ceil_div_i((long long)Nn * IN_F, T), T, 0, stream>>>(aggF, (size_t)Nn * IN_F);
    zero_f32_kernel<<<ceil_div_i(Nn, T), T, 0, stream>>>(deg, (size_t)Nn);
    scatter_f32_kernel<<<ceil_div_i((long long)E * 32, T), T, 0, stream>>>(
        x, src, dst, aggF, deg, E, IN_F);
    finalize_mean_kernel<<<ceil_div_i((long long)Nn * IN_F, T), T, 0, stream>>>(
        aggF, deg, aggB, (size_t)Nn * IN_F, IN_F);

    // 2) conv1 GEMM: h1 = relu(agg1@W1l^T + x@W1r^T + b1)   [Nn,128] bf16
    {
        int waves = (Nn >> 4) * (HIDF >> 6);
        gemm_bf16_wmma_kernel<<<ceil_div_i(waves, 8), 256, 0, stream>>>(
            aggB, w1lB, IN_F, xB, w1rB, IN_F, b1, Nn, HIDF, nullptr, h1B, 1);
    }

    // 3) conv2 aggregation: agg2 = segment_sum(h1[src], dst) (deg is identical)
    zero_f32_kernel<<<ceil_div_i((long long)Nn * HIDF, T), T, 0, stream>>>(aggF, (size_t)Nn * HIDF);
    scatter_bf16_kernel<<<ceil_div_i((long long)E * 32, T), T, 0, stream>>>(
        h1B, src, dst, aggF, E, HIDF);
    finalize_mean_kernel<<<ceil_div_i((long long)Nn * HIDF, T), T, 0, stream>>>(
        aggF, deg, aggB, (size_t)Nn * HIDF, HIDF);

    // 4) conv2 GEMM: h2 = relu(agg2@W2l^T + h1@W2r^T + b2)  [Nn,256] bf16
    {
        int waves = (Nn >> 4) * (OUTF >> 6);
        gemm_bf16_wmma_kernel<<<ceil_div_i(waves, 8), 256, 0, stream>>>(
            aggB, w2lB, HIDF, h1B, w2rB, HIDF, b2, Nn, OUTF, nullptr, h2B, 1);
    }

    // 5) projection: out = h2@Wp^T + bp   [Nn,384] f32 -> d_out
    {
        int waves = (Nn >> 4) * (FINF >> 6);
        gemm_bf16_wmma_kernel<<<ceil_div_i(waves, 8), 256, 0, stream>>>(
            h2B, wpB, OUTF, nullptr, nullptr, 0, bp, Nn, FINF, (float*)d_out, nullptr, 0);
    }
    (void)out_size;
}
